// CRFLayer_39376260170200
// MI455X (gfx1250) — matrix operand order repaired and checked by reading, not compiled, not run on hardware
//
#include <hip/hip_runtime.h>
#include <hip/hip_bf16.h>
#include <math.h>

typedef __attribute__((ext_vector_type(16))) _Float16 v16h;
typedef __attribute__((ext_vector_type(8)))  float    v8f;

#define TAGS      64
#define SEQ       1024
#define BATCH     512
#define NB        16            // batches per workgroup == WMMA N
#define START_TAG 62
#define STOP_TAG  63
#define NEG_INF   (-10000.0f)
#define FPAD      68            // padded floats per batch row in LDS feat tile
#define FTILE     (NB * FPAD)   // floats per double-buffer slot
#define LOG2E     1.4426950408889634f
#define LN2       0.6931471805599453f

// fast native base-2 transcendentals (v_exp_f32 / v_log_f32 are base-2 on AMDGPU)
__device__ __forceinline__ float fast_exp2(float x) {
#if __has_builtin(__builtin_amdgcn_exp2f)
  return __builtin_amdgcn_exp2f(x);
#else
  return exp2f(x);
#endif
}
__device__ __forceinline__ float fast_log2(float x) {
#if __has_builtin(__builtin_amdgcn_logf)
  return __builtin_amdgcn_logf(x);
#else
  return log2f(x);
#endif
}

// raw 32-bit LDS byte offset of a __shared__ object (for async-to-LDS dest VGPR)
__device__ __forceinline__ unsigned lds_byte_off(const void* p) {
  return (unsigned)(size_t)(__attribute__((address_space(3))) const void*)p;
}

// one wave-level async 16B/lane copy: global -> LDS (ASYNCcnt-tracked)
__device__ __forceinline__ void async_copy_b128(unsigned ldsByteOff, const float* gsrc) {
  asm volatile("global_load_async_to_lds_b128 %0, %1, off"
               :: "v"(ldsByteOff), "v"(gsrc)
               : "memory");
}

// ---------------------------------------------------------------------------
// Forward algorithm: one workgroup = 16 batches, 4 waves = 4 M-tiles of 16 tags.
// The forward variable is kept in the LOG2 domain (fv2 = fv * log2e), so the
// per-step transcendentals are raw v_exp_f32/v_log_f32 with no base-conversion
// fixup on the serial critical path.
// Each scan step is D(64x16) = expT(64x64,f16) x p(64x16,f16) via WMMA f32 accum.
// Emission tiles for step s+1 are staged into LDS with async-to-LDS copies.
// ---------------------------------------------------------------------------
__global__ __launch_bounds__(128)
void crf_forward_kernel(const float* __restrict__ feats,
                        const float* __restrict__ trans,
                        const int*   __restrict__ mask,
                        float* __restrict__ fwd_out) {
  __shared__ _Float16 sExpT[TAGS * TAGS];   // exp(trans - rowmax), f16, 8KB
  __shared__ v16h     sPBv[64];             // B fragments: 2 frags * 32 lanes * v16h
  __shared__ float    sFeat[2 * FTILE];     // double-buffered emission tiles
  __shared__ float    sFv[TAGS * 17];       // forward var (log2 domain), stride 17
  __shared__ float    sRmax2[TAGS];         // rowmax(trans) * log2e
  __shared__ float    sTstop2[TAGS];        // trans[STOP,:] * log2e
  __shared__ float    sPmax[8 * 16];
  __shared__ float    sM[16];

  const int tid   = threadIdx.x;
  const int lane  = tid & 31;
  const int wave  = tid >> 5;
  const int group = blockIdx.x;             // 32 groups of NB=16 batches

  const unsigned featBase = lds_byte_off(sFeat);

  // ---- kick off async copy of the s=0 emission tile into buffer 0 ----
  // tile = 16 batches x 64 floats; 256 x 16B chunks; 2 chunks per thread
  for (int k = 0; k < 2; ++k) {
    int chunk = k * 128 + tid;
    int cb = chunk >> 4;                    // batch within group
    int cc = chunk & 15;                    // 16B chunk within the 64-float row
    const float* g = feats + ((size_t)(group * NB + cb) * SEQ + 0) * TAGS + cc * 4;
    unsigned     l = featBase + (unsigned)(cb * FPAD + cc * 4) * 4u;
    async_copy_b128(l, g);
  }

  // ---- one-time preprocessing: row-max + exp(trans) into LDS (f16) ----
  if (tid < TAGS) {
    float mx = -3.4e38f;
    for (int j = 0; j < TAGS; ++j) mx = fmaxf(mx, trans[tid * TAGS + j]);
    sRmax2[tid]  = mx * LOG2E;
    sTstop2[tid] = trans[STOP_TAG * TAGS + tid] * LOG2E;
  }
  __syncthreads();
  for (int idx = tid; idx < TAGS * TAGS; idx += 128) {
    int i = idx >> 6;
    // exp(trans - rmax) == exp2((trans*log2e) - rmax2)
    sExpT[idx] = (_Float16)fast_exp2(trans[idx] * LOG2E - sRmax2[i]);
  }
  // init fv (log2 domain): NEG_INF*log2e except START tag
  for (int idx = tid; idx < TAGS * NB; idx += 128) {
    int j = idx >> 4, b = idx & 15;
    sFv[j * 17 + b] = (j == START_TAG) ? 0.f : NEG_INF * LOG2E;
  }
  __syncthreads();

  // ---- load constant A fragments (16x32 f16 each), ISA A-layout ----
  // lanes 0-15: M=lane, K = {0..7,16..23}; lanes 16-31: M=lane-16, K = {8..15,24..31}
  const int mo = wave * 16;
  const int hi = (lane >= 16) ? 1 : 0;
  v16h a0, a1;
  {
    const int M = mo + (lane & 15);
    for (int t = 0; t < 16; ++t) {
      int kk = ((t < 8) ? t : t + 8) + 8 * hi;
      a0[t] = sExpT[M * TAGS + kk];
      a1[t] = sExpT[M * TAGS + 32 + kk];
    }
  }

  const int bloc = tid & 15;   // batch column for reduction phases
  const int seg  = tid >> 4;   // 0..7 : 8-tag segment
  float regs[8];

  for (int s = 0; s < SEQ; ++s) {
    // issue async copy of step s+1's emission tile into the other buffer
    // (its previous readers finished before the end-of-step barrier of s-1)
    if (s + 1 < SEQ) {
      const int nbuf = (s + 1) & 1;
      for (int k = 0; k < 2; ++k) {
        int chunk = k * 128 + tid;
        int cb = chunk >> 4;
        int cc = chunk & 15;
        const float* g = feats + ((size_t)(group * NB + cb) * SEQ + (s + 1)) * TAGS + cc * 4;
        unsigned     l = featBase + (unsigned)(nbuf * FTILE + cb * FPAD + cc * 4) * 4u;
        async_copy_b128(l, g);
      }
    }

    // phase 1: per-batch max of fv2 (numerical stabilization)
    float mx = -3.4e38f;
    for (int jj = 0; jj < 8; ++jj) {
      float v = sFv[(seg * 8 + jj) * 17 + bloc];
      regs[jj] = v;
      mx = fmaxf(mx, v);
    }
    sPmax[seg * 16 + bloc] = mx;
    __syncthreads();
    if (tid < 16) {
      float m = sPmax[tid];
      for (int g = 1; g < 8; ++g) m = fmaxf(m, sPmax[g * 16 + tid]);
      sM[tid] = m;
    }
    __syncthreads();

    // phase 2: p = exp2(fv2 - m2) -> f16, stored directly in B-fragment layout:
    // frag f covers K=f*32..f*32+31; lane L holds N=L&15, K = {0..7,16..23} (+8 for L>=16)
    {
      float m = sM[bloc];
      _Float16* pbh = (_Float16*)sPBv;
      for (int jj = 0; jj < 8; ++jj) {
        int j  = seg * 8 + jj;
        int f  = j >> 5;
        int j5 = j & 31;
        int L  = bloc + (((j5 >> 3) & 1) ? 16 : 0);
        int t  = (j5 & 7) + ((j5 >= 16) ? 8 : 0);
        pbh[f * 512 + L * 16 + t] = (_Float16)fast_exp2(regs[jj] - m);
      }
    }
    // ensure emission tile for step s has landed (oldest 2 async ops per wave),
    // then make all waves' p-fragments AND async chunks visible
    if (s + 1 < SEQ) asm volatile("s_wait_asynccnt 0x2" ::: "memory");
    else             asm volatile("s_wait_asynccnt 0x0" ::: "memory");
    __syncthreads();

    // phase 3: D = expT * p  (K=64 -> two chained 16x16x32 f16 WMMAs), fv update
    v16h b0 = sPBv[lane];
    v16h b1 = sPBv[32 + lane];
    v8f  c  = {};
    c = __builtin_amdgcn_wmma_f32_16x16x32_f16(false, a0, false, b0, (short)0, c, false, false);
    c = __builtin_amdgcn_wmma_f32_16x16x32_f16(false, a1, false, b1, (short)0, c, false, false);

    const int    bcol  = lane & 15;                            // D-matrix N = batch
    const float  mB    = sM[bcol];
    const float  mf    = (float)mask[(size_t)(group * NB + bcol) * SEQ + s];
    const float* fRow  = &sFeat[(s & 1) * FTILE + bcol * FPAD + mo + hi * 8];
    for (int r = 0; r < 8; ++r) {
      int   M   = mo + r + hi * 8;                             // D-matrix M row
      float d   = fmaxf(c[r], 1e-30f);
      // fv2' = feat*log2e + rmax2 + m2 + log2(D)   (single fma + adds + raw v_log)
      float nxt = fmaf(fRow[r], LOG2E, sRmax2[M]) + mB + fast_log2(d);
      float old = sFv[M * 17 + bcol];
      sFv[M * 17 + bcol] = nxt * mf + old * (1.f - mf);
    }
    __syncthreads();
  }

  // ---- terminal: forward_score[b] = ln2 * log2sumexp2_i(fv2[i] + tstop2[i]) ----
  float tv[8];
  float tmx = -3.4e38f;
  for (int jj = 0; jj < 8; ++jj) {
    int j = seg * 8 + jj;
    tv[jj] = sFv[j * 17 + bloc] + sTstop2[j];
    tmx = fmaxf(tmx, tv[jj]);
  }
  sPmax[seg * 16 + bloc] = tmx;
  __syncthreads();
  if (tid < 16) {
    float m = sPmax[tid];
    for (int g = 1; g < 8; ++g) m = fmaxf(m, sPmax[g * 16 + tid]);
    sM[tid] = m;
  }
  __syncthreads();
  float m2 = sM[bloc];
  float ps = 0.f;
  for (int jj = 0; jj < 8; ++jj) ps += fast_exp2(tv[jj] - m2);
  __syncthreads();
  sPmax[seg * 16 + bloc] = ps;
  __syncthreads();
  if (tid < 16) {
    float sum = 0.f;
    for (int g = 0; g < 8; ++g) sum += sPmax[g * 16 + tid];
    fwd_out[group * NB + tid] = (sM[tid] + fast_log2(sum)) * LN2;
  }
}

// ---------------------------------------------------------------------------
// Gold score: per-batch gather/sum (memory-light).
// ---------------------------------------------------------------------------
__global__ __launch_bounds__(128)
void crf_gold_kernel(const float* __restrict__ feats,
                     const float* __restrict__ trans,
                     const int*   __restrict__ tags,
                     const int*   __restrict__ mask,
                     float* __restrict__ gold_out) {
  __shared__ float sRed[128];
  __shared__ int   sCnt[128];
  const int b   = blockIdx.x;
  const int tid = threadIdx.x;

  float acc = 0.f;
  int   cnt = 0;
  for (int s = tid; s < SEQ; s += 128) {
    int   cur  = tags[(size_t)b * SEQ + s];
    int   prev = (s == 0) ? START_TAG : tags[(size_t)b * SEQ + s - 1];
    int   m    = mask[(size_t)b * SEQ + s];
    float mf   = (float)m;
    acc += (feats[((size_t)b * SEQ + s) * TAGS + cur] + trans[cur * TAGS + prev]) * mf;
    cnt += m;
  }
  sRed[tid] = acc;
  sCnt[tid] = cnt;
  __syncthreads();
  for (int off = 64; off > 0; off >>= 1) {
    if (tid < off) {
      sRed[tid] += sRed[tid + off];
      sCnt[tid] += sCnt[tid + off];
    }
    __syncthreads();
  }
  if (tid == 0) {
    int lastIdx = sCnt[0];
    int lastTag = (lastIdx == 0) ? START_TAG : tags[(size_t)b * SEQ + lastIdx - 1];
    gold_out[b] = sRed[0] + trans[STOP_TAG * TAGS + lastTag];
  }
}

// ---------------------------------------------------------------------------
// Final mean of (forward - gold).
// ---------------------------------------------------------------------------
__global__ __launch_bounds__(256)
void crf_mean_kernel(const float* __restrict__ fwd,
                     const float* __restrict__ gold,
                     float* __restrict__ out) {
  __shared__ float sRed[256];
  const int tid = threadIdx.x;
  float a = 0.f;
  for (int i = tid; i < BATCH; i += 256) a += fwd[i] - gold[i];
  sRed[tid] = a;
  __syncthreads();
  for (int off = 128; off > 0; off >>= 1) {
    if (tid < off) sRed[tid] += sRed[tid + off];
    __syncthreads();
  }
  if (tid == 0) out[0] = sRed[0] * (1.0f / (float)BATCH);
}

extern "C" void kernel_launch(void* const* d_in, const int* in_sizes, int n_in,
                              void* d_out, int out_size, void* d_ws, size_t ws_size,
                              hipStream_t stream) {
  (void)in_sizes; (void)n_in; (void)out_size; (void)ws_size;
  const float* feats = (const float*)d_in[0];   // (512,1024,64) f32
  const float* trans = (const float*)d_in[1];   // (64,64) f32
  const int*   tags  = (const int*)d_in[2];     // (512,1024) int
  const int*   mask  = (const int*)d_in[3];     // (512,1024) int
  float* fwd  = (float*)d_ws;                   // 512 floats
  float* gold = fwd + BATCH;                    // 512 floats

  crf_forward_kernel<<<BATCH / NB, 128, 0, stream>>>(feats, trans, mask, fwd);
  crf_gold_kernel<<<BATCH, 128, 0, stream>>>(feats, trans, tags, mask, gold);
  crf_mean_kernel<<<1, 256, 0, stream>>>(fwd, gold, (float*)d_out);
}
